// SelfAttention_36326833390148
// MI455X (gfx1250) — compile-verified, matrix operand order checked by reading, not run on hardware
//
#include <hip/hip_runtime.h>
#include <hip/hip_bf16.h>
#include <math.h>

// Problem constants (fixed by the reference)
#define BB   4
#define SS   2048
#define EE   1024
#define HH   16
#define DD   64
#define E3   3072
#define MTOT (BB * SS)          // 8192 token rows

typedef __attribute__((ext_vector_type(16))) _Float16 v16h;
typedef __attribute__((ext_vector_type(8)))  _Float16 v8h;
typedef __attribute__((ext_vector_type(8)))  float    v8f;

// ---------------------------------------------------------------------------
// CDNA5 async global->LDS copy (GLOBAL_LOAD_ASYNC_TO_LDS_B128, ASYNCcnt),
// with a register-staged fallback if the builtin is unavailable.
// ---------------------------------------------------------------------------
#if __has_builtin(__builtin_amdgcn_global_load_async_to_lds_b128)
#define HAS_ASYNC 1
#else
#define HAS_ASYNC 0
#endif

typedef int v4i __attribute__((vector_size(16)));
typedef __attribute__((address_space(1))) v4i gv4i;
typedef __attribute__((address_space(3))) v4i lv4i;

static __device__ inline void async16(const _Float16* g, _Float16* l) {
#if HAS_ASYNC
    __builtin_amdgcn_global_load_async_to_lds_b128(
        (gv4i*)(void*)g, (lv4i*)(void*)l, 0, 0);
#else
    *(v8h*)l = *(const v8h*)g;
#endif
}

static __device__ inline void wait_async() {
#if HAS_ASYNC
#if __has_builtin(__builtin_amdgcn_s_wait_asynccnt)
    __builtin_amdgcn_s_wait_asynccnt(0);
#else
    asm volatile("s_wait_asynccnt 0x0" ::: "memory");
#endif
#endif
}

// raw v_exp_f32 (2^x); softmax runs in the base-2 domain
static __device__ inline float fast_exp2(float x) {
#if __has_builtin(__builtin_amdgcn_exp2f)
    return __builtin_amdgcn_exp2f(x);
#else
    return exp2f(x);
#endif
}

// Assemble a 16-half WMMA fragment from two contiguous 8-half (16B) chunks.
// Per CDNA5 ISA 16-bit A/B layout: lane L holds row (L&15); K-chunks at
// (L>>4)*8 and (L>>4)*8 + 16.
static __device__ inline v16h make_frag(const _Float16* p0, const _Float16* p1) {
    v8h lo = *(const v8h*)p0;
    v8h hi = *(const v8h*)p1;
    v16h r;
#pragma unroll
    for (int i = 0; i < 8; ++i) { r[i] = lo[i]; r[8 + i] = hi[i]; }
    return r;
}

// ---------------------------------------------------------------------------
// fp32 -> f16 conversion
// ---------------------------------------------------------------------------
__global__ void cvt_f32_f16(const float* __restrict__ in, _Float16* __restrict__ out, int n) {
    int i = blockIdx.x * blockDim.x + threadIdx.x;
    if (i < n) out[i] = (_Float16)in[i];
}

// bool padding mask -> additive float bias (0 or -1e30)
__global__ void mask_bias_kernel(const unsigned char* __restrict__ m,
                                 float* __restrict__ bias, int n) {
    int i = blockIdx.x * blockDim.x + threadIdx.x;
    if (i < n) bias[i] = m[i] ? -1.0e30f : 0.0f;
}

// ---------------------------------------------------------------------------
// Generic f16 GEMM:  out = (A[M x K] * W[N x K]^T + bias) * scale
// Block: 256 threads (8 waves, wave32), tile 128x256.  Wave tile: 64x64
// (16 WMMAs per K-step, A/B fragments each reused 4x).
// Double-buffered LDS staging via async-to-LDS; one barrier per K-step.
// MODE: 0 = f16 row-major, 1 = f16 (B,H,S,D) scatter, 2 = f16 (B,H,D,S)
//       transposed scatter, 3 = fp32 row-major
// ---------------------------------------------------------------------------
#define LDSW 40
template <int MODE>
__global__ __launch_bounds__(256) void gemm_f16(
    const _Float16* __restrict__ A, int lda,
    const _Float16* __restrict__ W, int ldb,
    const float* __restrict__ bias,
    void* __restrict__ out, int N, int K, float scale)
{
    __shared__ _Float16 As[2][128 * LDSW];
    __shared__ _Float16 Bs[2][256 * LDSW];

    const int tid  = threadIdx.x;
    const int lane = tid & 31;
    const int w    = tid >> 5;
    const int wm   = (w & 1) * 64;     // wave row offset in block tile
    const int wn   = (w >> 1) * 64;    // wave col offset in block tile
    const int mtile = blockIdx.y * 128;
    const int ntile = blockIdx.x * 256;
    const int l15 = lane & 15;
    const int g8  = (lane >> 4) * 8;

    v8f acc[4][4] = {};

    const int ldr = tid >> 1;            // 0..127 : row within tile
    const int ldc = (tid & 1) * 16;      // 0 or 16 : K chunk

    auto stage = [&](int buf, int k0s) {
        const _Float16* ga = A + (size_t)(mtile + ldr) * lda + k0s + ldc;
        async16(ga,     &As[buf][ldr * LDSW + ldc]);
        async16(ga + 8, &As[buf][ldr * LDSW + ldc + 8]);
        const _Float16* gb0 = W + (size_t)(ntile + ldr) * ldb + k0s + ldc;
        async16(gb0,     &Bs[buf][ldr * LDSW + ldc]);
        async16(gb0 + 8, &Bs[buf][ldr * LDSW + ldc + 8]);
        const _Float16* gb1 = W + (size_t)(ntile + 128 + ldr) * ldb + k0s + ldc;
        async16(gb1,     &Bs[buf][(128 + ldr) * LDSW + ldc]);
        async16(gb1 + 8, &Bs[buf][(128 + ldr) * LDSW + ldc + 8]);
    };

    stage(0, 0);
    wait_async();
    __syncthreads();

    int ib = 0;
    for (int k0 = 0; k0 < K; k0 += 32, ib ^= 1) {
        if (k0 + 32 < K) stage(ib ^ 1, k0 + 32);   // prefetch next tile

        v16h bf[4];
#pragma unroll
        for (int j = 0; j < 4; ++j) {
            int row = wn + j * 16 + l15;
            bf[j] = make_frag(&Bs[ib][row * LDSW + g8], &Bs[ib][row * LDSW + g8 + 16]);
        }
#pragma unroll
        for (int i = 0; i < 4; ++i) {
            int row = wm + i * 16 + l15;
            v16h af = make_frag(&As[ib][row * LDSW + g8], &As[ib][row * LDSW + g8 + 16]);
#pragma unroll
            for (int j = 0; j < 4; ++j) {
                acc[i][j] = __builtin_amdgcn_wmma_f32_16x16x32_f16(
                    false, af, false, bf[j], (short)0, acc[i][j], false, false);
            }
        }
        wait_async();          // next tile landed
        __syncthreads();       // everyone done reading current tile
    }

    // store: C layout -> lanes 0-15: M = reg, lanes 16-31: M = reg + 8; N = lane&15
#pragma unroll
    for (int i = 0; i < 4; ++i) {
#pragma unroll
        for (int j = 0; j < 4; ++j) {
            int n = ntile + wn + j * 16 + l15;
            float bv = bias ? bias[n] : 0.0f;
#pragma unroll
            for (int r = 0; r < 8; ++r) {
                int m = mtile + wm + i * 16 + g8 + r;
                float val = (acc[i][j][r] + bv) * scale;
                if (MODE == 0) {
                    ((_Float16*)out)[(size_t)m * N + n] = (_Float16)val;
                } else if (MODE == 3) {
                    ((float*)out)[(size_t)m * N + n] = val;
                } else {
                    int b = m >> 11, s = m & (SS - 1);
                    int h = n >> 6,  d = n & (DD - 1);
                    if (MODE == 1) {
                        ((_Float16*)out)[(((size_t)(b * HH + h) * SS) + s) * DD + d] = (_Float16)val;
                    } else { // MODE 2: V transposed (B,H,D,S)
                        ((_Float16*)out)[(((size_t)(b * HH + h) * DD) + d) * SS + s] = (_Float16)val;
                    }
                }
            }
        }
    }
}

// ---------------------------------------------------------------------------
// RoPE (rotate-half) applied in place to the f16 combined (M x 3E) buffer,
// q = cols [0,1024), k = cols [1024,2048).
// ---------------------------------------------------------------------------
__global__ void rope_kernel(_Float16* __restrict__ C) {
    int row = blockIdx.x;                 // 0..8191
    int s = row & (SS - 1);
    size_t base = (size_t)row * E3;
    for (int p = threadIdx.x; p < 1024; p += blockDim.x) {
        int qk  = p >> 9;                 // 0 = q, 1 = k
        int idx = p & 511;
        int h = idx >> 5, j = idx & 31;
        int col0 = qk * EE + h * DD + j;
        int col1 = col0 + 32;
        // inv_freq = 10000^(-j/32) ;  ln(10000) = 9.210340371976184
        float invf = expf(-(float)j * (9.210340371976184f / 32.0f));
        float ang = (float)s * invf;
        float c = cosf(ang), sn = sinf(ang);
        float t0 = (float)C[base + col0];
        float t1 = (float)C[base + col1];
        C[base + col0] = (_Float16)(t0 * c - t1 * sn);
        C[base + col1] = (_Float16)(t1 * c + t0 * sn);
    }
}

// ---------------------------------------------------------------------------
// Flash attention. Grid: (S/128, B*H). 8 waves/block, 16 queries/wave.
// K/V^T tiles double-buffered in LDS (async-to-LDS prefetch of block i+1
// overlaps the WMMAs of block i; one barrier per iteration).
// Computes S^T = K·Q^T so softmax stats are lane-local (+ one shfl_xor 16),
// and the S^T C-fragment converts register-for-register into the P^T
// B-fragment.  Then O^T += V^T · P^T.
// Q is pre-scaled by log2(e)/sqrt(D): the whole softmax runs in base-2 and
// every exponential is a single v_exp_f32.
// ---------------------------------------------------------------------------
#define KSW 72   // K-tile LDS row stride (32 rows x 64 halves, padded)
#define VSW 40   // V^T-tile LDS row stride (64 rows x 32 halves, padded)
__global__ __launch_bounds__(256) void attn_kernel(
    const _Float16* __restrict__ Q,   // (B,H,S,D) pre-scaled (incl. log2 e)
    const _Float16* __restrict__ Kh,  // (B,H,S,D)
    const _Float16* __restrict__ VT,  // (B,H,D,S)
    const float* __restrict__ mbias,  // (B,S) additive mask bias
    _Float16* __restrict__ ctx)       // (B,S,H,D) == (B,S,E)
{
    __shared__ _Float16 Ks[2][32 * KSW];
    __shared__ _Float16 Vs[2][64 * VSW];

    const int tid  = threadIdx.x;
    const int lane = tid & 31;
    const int w    = tid >> 5;
    const int bh = blockIdx.y;
    const int b  = bh >> 4;           // /HH
    const int h  = bh & (HH - 1);
    const int q0 = blockIdx.x * 128 + w * 16;
    const int l15 = lane & 15;
    const int g8  = (lane >> 4) * 8;

    const _Float16* qbase = Q  + ((size_t)bh * SS + q0) * DD;
    const _Float16* kbase = Kh + (size_t)bh * SS * DD;
    const _Float16* vbase = VT + (size_t)bh * DD * SS;
    const float*    mb    = mbias + (size_t)b * SS;

    // staging assignments (per thread, one 16B chunk per tile)
    const int kr = tid >> 3, kc = (tid & 7) * 8;   // K tile: 32 x 64
    const int vr = tid >> 2, vc = (tid & 3) * 8;   // V^T tile: 64 x 32

    auto stage = [&](int buf, int kbs) {
        async16(kbase + (size_t)(kbs + kr) * DD + kc, &Ks[buf][kr * KSW + kc]);
        async16(vbase + (size_t)vr * SS + kbs + vc,   &Vs[buf][vr * VSW + vc]);
    };

    // Q fragments (used as the B operand of S^T = K·Q^T)
    v16h qf[2];
#pragma unroll
    for (int f = 0; f < 2; ++f) {
        const _Float16* p = qbase + (size_t)l15 * DD + f * 32 + g8;
        qf[f] = make_frag(p, p + 16);
    }

    v8f o[4] = {};                 // O^T accum: dims (4x16) x queries (16)
    float mrun = -3.0e38f, lrun = 0.0f;

    stage(0, 0);
    wait_async();
    __syncthreads();

    int ib = 0;
    for (int kb = 0; kb < SS; kb += 32, ib ^= 1) {
        if (kb + 32 < SS) stage(ib ^ 1, kb + 32);   // prefetch next key block

        // S^T fragments: c0 = keys kb..kb+15, c1 = keys kb+16..kb+31
        v8f c0 = {}, c1 = {};
#pragma unroll
        for (int f = 0; f < 2; ++f) {
            v16h kf0 = make_frag(&Ks[ib][l15 * KSW + f * 32 + g8],
                                 &Ks[ib][l15 * KSW + f * 32 + g8 + 16]);
            c0 = __builtin_amdgcn_wmma_f32_16x16x32_f16(
                false, kf0, false, qf[f], (short)0, c0, false, false);
            v16h kf1 = make_frag(&Ks[ib][(16 + l15) * KSW + f * 32 + g8],
                                 &Ks[ib][(16 + l15) * KSW + f * 32 + g8 + 16]);
            c1 = __builtin_amdgcn_wmma_f32_16x16x32_f16(
                false, kf1, false, qf[f], (short)0, c1, false, false);
        }

        // additive padding-mask bias (aligned 32B vector loads)
        v8f mv0 = *(const v8f*)&mb[kb + g8];
        v8f mv1 = *(const v8f*)&mb[kb + 16 + g8];
#pragma unroll
        for (int r = 0; r < 8; ++r) { c0[r] += mv0[r]; c1[r] += mv1[r]; }

        // online softmax (base-2) per query (lane pair n / n+16 share query n)
        float tmax = c0[0];
#pragma unroll
        for (int r = 1; r < 8; ++r) tmax = fmaxf(tmax, c0[r]);
#pragma unroll
        for (int r = 0; r < 8; ++r) tmax = fmaxf(tmax, c1[r]);
        tmax = fmaxf(tmax, __shfl_xor(tmax, 16, 32));
        float mnew = fmaxf(mrun, tmax);
        float corr = fast_exp2(mrun - mnew);

        float lsum = 0.0f;
        v16h pt;                      // P^T B-fragment (reg-for-reg from c0/c1)
#pragma unroll
        for (int r = 0; r < 8; ++r) {
            float p = fast_exp2(c0[r] - mnew); lsum += p; pt[r] = (_Float16)p;
        }
#pragma unroll
        for (int r = 0; r < 8; ++r) {
            float p = fast_exp2(c1[r] - mnew); lsum += p; pt[8 + r] = (_Float16)p;
        }
        lsum += __shfl_xor(lsum, 16, 32);
        lrun = lrun * corr + lsum;
        mrun = mnew;

#pragma unroll
        for (int d = 0; d < 4; ++d)
#pragma unroll
            for (int r = 0; r < 8; ++r) o[d][r] *= corr;

        // O^T += V^T · P^T   (A = V^T dims x keys)
#pragma unroll
        for (int d = 0; d < 4; ++d) {
            v16h vf = make_frag(&Vs[ib][(d * 16 + l15) * VSW + g8],
                                &Vs[ib][(d * 16 + l15) * VSW + g8 + 16]);
            o[d] = __builtin_amdgcn_wmma_f32_16x16x32_f16(
                false, vf, false, pt, (short)0, o[d], false, false);
        }

        wait_async();        // next tiles landed
        __syncthreads();     // everyone done reading current tiles
    }

    const float inv = 1.0f / lrun;
#pragma unroll
    for (int d = 0; d < 4; ++d) {
        v8h st;
#pragma unroll
        for (int r = 0; r < 8; ++r) st[r] = (_Float16)(o[d][r] * inv);
        // query = q0 + l15 ; dims = d*16 + g8 .. +7 (contiguous -> 16B store)
        size_t off = (((size_t)b * SS + q0 + l15) * HH + h) * DD + d * 16 + g8;
        *(v8h*)&ctx[off] = st;
    }
}

// ---------------------------------------------------------------------------
// Host-side orchestration
// ---------------------------------------------------------------------------
extern "C" void kernel_launch(void* const* d_in, const int* in_sizes, int n_in,
                              void* d_out, int out_size, void* d_ws, size_t ws_size,
                              hipStream_t stream) {
    const float* x         = (const float*)d_in[0];          // (B,S,E)
    const float* in_proj_w = (const float*)d_in[1];          // (3E,E)
    const float* in_proj_b = (const float*)d_in[2];          // (3E,)
    const float* out_w     = (const float*)d_in[3];          // (E,E)
    const float* out_b     = (const float*)d_in[4];          // (E,)
    const unsigned char* pmask = (const unsigned char*)d_in[5]; // (B,S) bool
    float* out = (float*)d_out;

    // workspace layout (f16 elements)
    _Float16* Wh  = (_Float16*)d_ws;            // in_proj_w f16     : 3E*E
    _Float16* Oh  = Wh  + (size_t)E3 * EE;      // out_w f16         : E*E
    _Float16* Xh  = Oh  + (size_t)EE * EE;      // x f16             : M*E
    _Float16* Ch  = Xh  + (size_t)MTOT * EE;    // combined f16      : M*3E
    _Float16* QH  = Ch  + (size_t)MTOT * E3;    // (B,H,S,D) scaled
    _Float16* KH  = QH  + (size_t)MTOT * EE;    // (B,H,S,D)
    _Float16* VT  = KH  + (size_t)MTOT * EE;    // (B,H,D,S)
    _Float16* CTX = VT  + (size_t)MTOT * EE;    // (B,S,E)
    float*    MB  = (float*)(CTX + (size_t)MTOT * EE);  // (B,S) mask bias

    const int nW = E3 * EE, nO = EE * EE, nX = MTOT * EE;
    cvt_f32_f16<<<(nW + 255) / 256, 256, 0, stream>>>(in_proj_w, Wh, nW);
    cvt_f32_f16<<<(nO + 255) / 256, 256, 0, stream>>>(out_w, Oh, nO);
    cvt_f32_f16<<<(nX + 255) / 256, 256, 0, stream>>>(x, Xh, nX);
    mask_bias_kernel<<<(BB * SS + 255) / 256, 256, 0, stream>>>(pmask, MB, BB * SS);

    // 1) combined = x @ in_proj_w^T + b     (8192 x 3072)
    gemm_f16<0><<<dim3(E3 / 256, MTOT / 128), 256, 0, stream>>>(
        Xh, EE, Wh, EE, in_proj_b, Ch, E3, EE, 1.0f);

    // 2) RoPE on q,k halves of combined (in place)
    rope_kernel<<<MTOT, 256, 0, stream>>>(Ch);

    // 3) second projection: q2 (scaled by log2(e)/8), k2, v2 (v transposed)
    gemm_f16<1><<<dim3(EE / 256, MTOT / 128), 256, 0, stream>>>(
        Ch, E3, Wh, EE, in_proj_b, QH, EE, EE, 0.125f * 1.4426950408889634f);
    gemm_f16<1><<<dim3(EE / 256, MTOT / 128), 256, 0, stream>>>(
        Ch + EE, E3, Wh + (size_t)EE * EE, EE, in_proj_b + EE, KH, EE, EE, 1.0f);
    gemm_f16<2><<<dim3(EE / 256, MTOT / 128), 256, 0, stream>>>(
        Ch + 2 * EE, E3, Wh + 2 * (size_t)EE * EE, EE, in_proj_b + 2 * EE, VT, EE, EE, 1.0f);

    // 4) flash attention -> ctx (B,S,E) f16
    attn_kernel<<<dim3(SS / 128, BB * HH), 256, 0, stream>>>(QH, KH, VT, MB, CTX);

    // 5) out = ctx @ out_w^T + out_b   (fp32)
    gemm_f16<3><<<dim3(EE / 256, MTOT / 128), 256, 0, stream>>>(
        CTX, EE, Oh, EE, out_b, out, EE, EE, 1.0f);
}
